// AdaConv_37151467111008
// MI455X (gfx1250) — matrix operand order repaired and checked by reading, not compiled, not run on hardware
//
#include <hip/hip_runtime.h>

#define N_S   8
#define C_INC 512
#define SD    512
#define HH    128
#define WW    128
#define NG    8
#define CPG   64          // channels per group
#define KTOT  576         // 64 ic * 9 taps
#define OC_ROWS 32768     // C_OUT * CIN_PG = dw predictor rows
#define NGT   (N_S * NG)  // 64 (n,g) pairs
#define WFRAG_DW 18432    // per-(n,g) W_comb fragment dwords: 4*9*2*256
#define LXS_DW  (4 * 130 * 33)   // conv LDS tile dwords (17160) = 68,640 B

typedef __attribute__((ext_vector_type(16))) __bf16 v16bf;
typedef __attribute__((ext_vector_type(8)))  float  v8f;

union Frag { unsigned int u[8]; v16bf v; };

// hardware f32->bf16 (RNE); pairs lower to v_cvt_pk_bf16_f32
__device__ __forceinline__ unsigned short f2bfu(float f) {
  __bf16 b = (__bf16)f;
  return __builtin_bit_cast(unsigned short, b);
}
__device__ __forceinline__ unsigned int pk2bf(float lo, float hi) {
  return (unsigned int)f2bfu(lo) | ((unsigned int)f2bfu(hi) << 16);
}

// ---------------- 1a. per-sample sum/sumsq partials (deterministic) ---------
__global__ void k_stats(const float* __restrict__ x, float* __restrict__ partial) {
  const int n = blockIdx.x >> 8;
  const int b = blockIdx.x & 255;
  const size_t per = (size_t)C_INC * HH * WW;          // 8388608
  const float* xb = x + (size_t)n * per + (size_t)b * (per / 256);
  float s = 0.f, s2 = 0.f;
  for (int i = threadIdx.x; i < (int)(per / 256); i += 256) {
    float v = xb[i];
    s += v; s2 += v * v;
  }
  __shared__ float sh[256], sh2[256];
  sh[threadIdx.x] = s; sh2[threadIdx.x] = s2; __syncthreads();
  for (int o = 128; o > 0; o >>= 1) {
    if (threadIdx.x < o) { sh[threadIdx.x] += sh[threadIdx.x + o]; sh2[threadIdx.x] += sh2[threadIdx.x + o]; }
    __syncthreads();
  }
  if (threadIdx.x == 0) {
    partial[(n * 256 + b) * 2]     = sh[0];
    partial[(n * 256 + b) * 2 + 1] = sh2[0];
  }
}

// ---------------- 1b. finalize mean / inv_std -------------------------------
__global__ void k_stats_final(const float* __restrict__ partial,
                              const unsigned char* __restrict__ normflag,
                              float* __restrict__ meaninv) {
  const int n = blockIdx.x, t = threadIdx.x;
  __shared__ float sh[256], sh2[256];
  sh[t]  = partial[(n * 256 + t) * 2];
  sh2[t] = partial[(n * 256 + t) * 2 + 1];
  __syncthreads();
  for (int o = 128; o > 0; o >>= 1) {
    if (t < o) { sh[t] += sh[t + o]; sh2[t] += sh2[t + o]; }
    __syncthreads();
  }
  if (t == 0) {
    const float invper = 1.0f / 8388608.0f;
    float mean = sh[0] * invper;
    float var  = sh2[0] * invper - mean * mean;
    float inv  = rsqrtf(var + 1e-5f);
    if (*normflag == 0) { mean = 0.f; inv = 1.f; }
    meaninv[n * 2] = mean; meaninv[n * 2 + 1] = inv;
  }
}

// ---------------- 2. style global mean (AvgPool 4x4) ------------------------
__global__ void k_style_mean(const float* __restrict__ style, float* __restrict__ smean) {
  int idx = blockIdx.x * 256 + threadIdx.x;            // < N_S*SD = 4096
  const float* p = style + (size_t)idx * 16;
  float s = 0.f;
#pragma unroll
  for (int i = 0; i < 16; ++i) s += p[i];
  smean[idx] = s * (1.f / 16.f);
}

// ---------------- 2b. pre-pack style patches as bf16 B-matrix ---------------
// bsty[col][kd] dwords, col = n*9+p (72 cols), k = ic*4 + kh*2 + kw, kd = ic*2+kh
__global__ void k_bsty(const float* __restrict__ style, unsigned int* __restrict__ bsty) {
  int idx = blockIdx.x * 256 + threadIdx.x;            // < 72*1024 = 73728
  int col = idx >> 10;
  int kd  = idx & 1023;
  int n = col / 9, p = col % 9;
  int ph = p / 3, pw = p % 3;
  int ic = kd >> 1, kh = kd & 1;
  const float* sp = style + (((size_t)n * SD + ic) * 4 + ph + kh) * 4 + pw;
  bsty[idx] = pk2bf(sp[0], sp[1]);                     // kw = 0,1 contiguous
}

// ---------------- 3. depthwise-kernel predictor GEMM (bf16 WMMA) ------------
// M = 32768 (oc), cols = 72 (5 tiles of 16), K = 2048 (64 chunks of 32)
__global__ void k_dw_gemm(const unsigned int* __restrict__ bsty,
                          const float* __restrict__ dkw,
                          const float* __restrict__ dkb,
                          float* __restrict__ dw) {
  const int wave = blockIdx.x * 8 + (threadIdx.x >> 5);    // oc-tile 0..2047
  const int lane = threadIdx.x & 31;
  const int m16  = lane & 15;
  const int hi   = lane >> 4;

  int coln[5], colp[5], colv[5], bofs[5];
#pragma unroll
  for (int ct = 0; ct < 5; ++ct) {
    int col = ct * 16 + m16;
    colv[ct] = (col < 72);
    int cc = colv[ct] ? col : 0;
    coln[ct] = cc / 9;
    colp[ct] = cc % 9;
    bofs[ct] = cc << 10;                               // col * 1024 dwords
  }

  v8f zero = {0.f,0.f,0.f,0.f,0.f,0.f,0.f,0.f};
  v8f acc[5];
#pragma unroll
  for (int ct = 0; ct < 5; ++ct) acc[ct] = zero;

  const float* arow = dkw + (size_t)(wave * 16 + m16) * 2048;   // K order == memory order

  for (int chunk = 0; chunk < 64; ++chunk) {
    Frag a;
#pragma unroll
    for (int v = 0; v < 8; ++v) {              // A 16x32 bf16 layout
      int kl = hi * 16 + 2 * v;
      const float* p = arow + chunk * 32 + kl;
      a.u[v] = pk2bf(p[0], p[1]);
    }
#pragma unroll
    for (int ct = 0; ct < 5; ++ct) {
      Frag b;
      const unsigned int* bp = bsty + bofs[ct] + chunk * 16 + hi * 8;
#pragma unroll
      for (int v = 0; v < 8; ++v) b.u[v] = bp[v];      // 8 consecutive dwords
      acc[ct] = __builtin_amdgcn_wmma_f32_16x16x32_bf16(false, a.v, false, b.v,
                                                        (short)0, acc[ct], false, false);
    }
  }

#pragma unroll
  for (int ct = 0; ct < 5; ++ct) {
    if (!colv[ct]) continue;
#pragma unroll
    for (int v = 0; v < 8; ++v) {
      int oc = wave * 16 + hi * 8 + v;
      dw[((size_t)coln[ct] * OC_ROWS + oc) * 9 + colp[ct]] = acc[ct][v] + dkb[oc];
    }
  }
}

// ---------------- 4. pointwise-kernel predictor (VALU, w streamed once) -----
__global__ void k_pk(const float* __restrict__ pwkw, const float* __restrict__ pwkb,
                     const float* __restrict__ smean, float* __restrict__ pk) {
  int r = blockIdx.x * 256 + threadIdx.x;              // < 32768
  const float* wrow = pwkw + (size_t)r * SD;
  float acc[N_S];
#pragma unroll
  for (int n = 0; n < N_S; ++n) acc[n] = 0.f;
  for (int ic = 0; ic < SD; ++ic) {
    float w = wrow[ic];
#pragma unroll
    for (int n = 0; n < N_S; ++n) acc[n] += w * smean[n * SD + ic];
  }
  float bias = pwkb[r];
#pragma unroll
  for (int n = 0; n < N_S; ++n) pk[(size_t)n * OC_ROWS + r] = acc[n] + bias;
}

// ---------------- 5. pointwise bias predictor -------------------------------
__global__ void k_pb(const float* __restrict__ pwbw, const float* __restrict__ pwbb,
                     const float* __restrict__ smean, float* __restrict__ pb) {
  int r = blockIdx.x * 256 + threadIdx.x;
  if (r >= C_INC) return;
  const float* wrow = pwbw + (size_t)r * SD;
  float acc[N_S];
#pragma unroll
  for (int n = 0; n < N_S; ++n) acc[n] = 0.f;
  for (int ic = 0; ic < SD; ++ic) {
    float w = wrow[ic];
#pragma unroll
    for (int n = 0; n < N_S; ++n) acc[n] += w * smean[n * SD + ic];
  }
#pragma unroll
  for (int n = 0; n < N_S; ++n) pb[n * C_INC + r] = acc[n] + pwbb[r];
}

// ---------------- 6. fuse PK @ DW -> W_comb in A-fragment order -------------
// fragment base = ((ng*4 + t)*9 + s)*2 + icc ; within: dword = lane*8 + v
// (lane = hi*16 + m16) -> each lane reads 2x b128, wave reads 1KB contiguous
__global__ void k_combine(const float* __restrict__ pk, const float* __restrict__ dw,
                          unsigned short* __restrict__ wc) {
  const int n = blockIdx.x >> 3, g = blockIdx.x & 7;
  const int ng = blockIdx.x;
  __shared__ float shpk[64 * 64];
  for (int i = threadIdx.x; i < 4096; i += 256)
    shpk[i] = pk[(size_t)n * OC_ROWS + (size_t)(g * 64 + (i >> 6)) * 64 + (i & 63)];
  __syncthreads();

  for (int idx = threadIdx.x; idx < 64 * KTOT; idx += 256) {
    int oc2 = idx / KTOT;
    int k   = idx % KTOT;
    int s   = k >> 6;              // tap kh*3+kw
    int i   = k & 63;              // local input channel
    // dw[n][c=g*64+m][i][s], stride over m = 576 floats
    const float* dwp = dw + ((size_t)n * OC_ROWS + (size_t)g * 64 * 64 + i) * 9 + s;
    const float* pkp = shpk + oc2 * 64;
    float acc = 0.f;
    for (int m = 0; m < 64; ++m) acc += pkp[m] * dwp[(size_t)m * 576];

    // map (oc2, s, i) -> fragment position
    int t   = oc2 >> 4;
    int m16 = oc2 & 15;
    int icc = i >> 5;
    int kl  = i & 31;
    int hi, v, bsel;
    hi = kl >> 4; v = (kl & 15) >> 1; bsel = kl & 1;
    size_t dwidx = ((((size_t)ng * 4 + t) * 9 + s) * 2 + icc) * 256
                 + (size_t)(hi * 16 + m16) * 8 + v;    // per-lane contiguous
    wc[dwidx * 2 + bsel] = f2bfu(acc);
  }
}

// ---------------- 7. fused norm + reflect-pad + grouped 3x3 conv (WMMA) -----
// One block = (n, g, 2 output rows). Each A fragment feeds 2 WMMAs (row0,row1).
__global__ void k_conv(const float* __restrict__ x,
                       const unsigned int* __restrict__ wc,
                       const float* __restrict__ pb,
                       const float* __restrict__ meaninv,
                       float* __restrict__ out) {
  const int hp = blockIdx.x & 63;                      // row pair
  const int ng = blockIdx.x >> 6;
  const int g  = ng & 7;
  const int n  = ng >> 3;
  const int h0 = hp * 2;

  // [padded row 0..3][col 0..129][icd 0..31], stride 33 dwords (conflict-free)
  extern __shared__ unsigned int lxs[];                // 4*130*33 dw = 68,640 B

  const float mean = meaninv[n * 2];
  const float inv  = meaninv[n * 2 + 1];
  const float* xg = x + ((size_t)n * C_INC + g * CPG) * (HH * WW);

  // pixels innermost across threads -> coalesced global reads
  for (int idx = threadIdx.x; idx < 4 * 130 * 32; idx += 256) {
    int t   = idx % 520;                               // 4*130 pixels
    int icd = idx / 520;
    int pc = t % 130;
    int pr = t / 130;
    int r = h0 - 1 + pr;  r = (r < 0) ? -r : ((r >= HH) ? (2 * HH - 2 - r) : r);
    int c = pc - 1;       c = (c < 0) ? -c : ((c >= WW) ? (2 * WW - 2 - c) : c);
    const float* p0 = xg + ((size_t)(2 * icd) * HH + r) * WW + c;
    float lo = (p0[0] - mean) * inv;
    float hi = (p0[HH * WW] - mean) * inv;
    lxs[t * 33 + icd] = pk2bf(lo, hi);
  }
  __syncthreads();

  const int lane = threadIdx.x & 31;
  const int m16  = lane & 15;
  const int hi   = lane >> 4;
  const int px   = (threadIdx.x >> 5) * 16 + m16;      // output pixel (B column)

  const unsigned int* wf = wc + (size_t)ng * WFRAG_DW;

  v8f zero = {0.f,0.f,0.f,0.f,0.f,0.f,0.f,0.f};
  v8f acc[2][4];
#pragma unroll
  for (int rr = 0; rr < 2; ++rr)
#pragma unroll
    for (int t = 0; t < 4; ++t) acc[rr][t] = zero;

#pragma unroll
  for (int s = 0; s < 9; ++s) {                        // 9 taps
    const int kh = s / 3, kw = s % 3;
#pragma unroll
    for (int icc = 0; icc < 2; ++icc) {                // two 32-ic halves
      Frag b0, b1;                                     // rows h0, h0+1; shared over t
      const unsigned int* bp = lxs + (kh * 130 + px + kw) * 33 + icc * 16 + hi * 8;
#pragma unroll
      for (int v = 0; v < 8; ++v) { b0.u[v] = bp[v]; b1.u[v] = bp[130 * 33 + v]; }
#pragma unroll
      for (int t = 0; t < 4; ++t) {
        const unsigned int* ap = wf + (((size_t)t * 9 + s) * 2 + icc) * 256 + lane * 8;
        Frag a;
#pragma unroll
        for (int v = 0; v < 8; ++v) a.u[v] = ap[v];    // 2x b128 per lane, 1KB/wave
        acc[0][t] = __builtin_amdgcn_wmma_f32_16x16x32_bf16(false, a.v, false, b0.v,
                                                            (short)0, acc[0][t], false, false);
        acc[1][t] = __builtin_amdgcn_wmma_f32_16x16x32_bf16(false, a.v, false, b1.v,
                                                            (short)0, acc[1][t], false, false);
      }
    }
  }

#pragma unroll
  for (int rr = 0; rr < 2; ++rr) {
#pragma unroll
    for (int t = 0; t < 4; ++t) {
#pragma unroll
      for (int v = 0; v < 8; ++v) {
        int oc = t * 16 + hi * 8 + v;
        float bias = pb[(size_t)n * C_INC + g * CPG + oc];
        out[(((size_t)n * C_INC + g * CPG + oc) * HH + (h0 + rr)) * WW + px] =
            acc[rr][t][v] + bias;
      }
    }
  }
}

// ---------------------------------------------------------------------------
extern "C" void kernel_launch(void* const* d_in, const int* in_sizes, int n_in,
                              void* d_out, int out_size, void* d_ws, size_t ws_size,
                              hipStream_t stream) {
  (void)in_sizes; (void)n_in; (void)out_size; (void)ws_size;
  const float* style = (const float*)d_in[0];
  const float* pred  = (const float*)d_in[1];
  const float* dkw   = (const float*)d_in[2];
  const float* dkb   = (const float*)d_in[3];
  const float* pwkw  = (const float*)d_in[4];
  const float* pwkb  = (const float*)d_in[5];
  const float* pwbw  = (const float*)d_in[6];
  const float* pwbb  = (const float*)d_in[7];
  const unsigned char* norm = (const unsigned char*)d_in[8];
  float* out = (float*)d_out;

  // workspace layout (~15.6 MB)
  float* partial = (float*)d_ws;                       // 4096 f
  float* meaninv = partial + 4096;                     // 16 f
  float* smean   = meaninv + 16;                       // 4096 f
  float* dw      = smean + 4096;                       // 2359296 f
  float* pk      = dw + 2359296;                       // 262144 f
  float* pb      = pk + 262144;                        // 4096 f
  unsigned int* bsty = (unsigned int*)(pb + 4096);     // 73728 dw (style bf16 B)
  unsigned int* wcomb = bsty + 73728;                  // 1179648 dw (W_comb frags)

  k_stats      <<<N_S * 256,       256, 0, stream>>>(pred, partial);
  k_stats_final<<<N_S,             256, 0, stream>>>(partial, norm, meaninv);
  k_style_mean <<<16,              256, 0, stream>>>(style, smean);
  k_bsty       <<<288,             256, 0, stream>>>(style, bsty);
  k_dw_gemm    <<<256,             256, 0, stream>>>(bsty, dkw, dkb, dw);
  k_pk         <<<128,             256, 0, stream>>>(pwkw, pwkb, smean, pk);
  k_pb         <<<2,               256, 0, stream>>>(pwbw, pwbb, smean, pb);
  k_combine    <<<N_S * NG,        256, 0, stream>>>(pk, dw, (unsigned short*)wcomb);
  k_conv       <<<N_S * NG * (HH/2), 256, LXS_DW * 4, stream>>>(pred, wcomb, pb, meaninv, out);
}
